// Block_25142738551093
// MI455X (gfx1250) — matrix operand order repaired and checked by reading, not compiled, not run on hardware
//
#include <hip/hip_runtime.h>
#include <math.h>

// ---------------------------------------------------------------------------
// Problem constants (fixed by the reference module)
// ---------------------------------------------------------------------------
#define BB     2
#define HIMG   128
#define WIMG   128
#define NPIX   16384           // HIMG*WIMG = 2^14
#define CIN    256
#define HEADS  8
#define DD     8
#define SD     126
#define MPIX   (BB*NPIX)       // 32768 pixel rows
#define MROWS  (BB*HEADS*NPIX) // 262144 (b,h,p) rows

typedef __attribute__((ext_vector_type(16))) _Float16 v16h;
typedef __attribute__((ext_vector_type(8)))  _Float16 v8h;
typedef __attribute__((ext_vector_type(8)))  float    v8f;

// ---------------------------------------------------------------------------
// Generic WMMA GEMM with A-fragment reuse: one wave computes 16 x (16*NSUB).
//   A: f16 row-major (M x lda), M % 16 == 0, K % 32 == 0, lda % 8 == 0
//   Bt: f16 row-major (Npad x ldb), Npad covers all subtiles, pads zeroed.
//   Fragment layout (ISA 7.12.2), lane l (r=l&15, hi=l>>4):
//     A 16x32: row M=r, K in {hi*8..+7} and {16+hi*8..+7}  -> 2 x b128
//     B 32x16: col N=r, K in {hi*8..+7} and {16+hi*8..+7}  -> 2 x b128
//     C/D v8f elem i -> row 8*hi+i, col r
// ---------------------------------------------------------------------------
template <int NSUB>
__global__ __launch_bounds__(32)
void wmma_gemm_f16(const _Float16* __restrict__ A, int lda,
                   const _Float16* __restrict__ Bt, int ldb,
                   const float* __restrict__ bias,
                   const float* __restrict__ aux, int ldaux, int auxoff,
                   float* __restrict__ OutF, int ldof,
                   _Float16* __restrict__ OutH, int ldoh,
                   int N, int K)
{
  const int lane = threadIdx.x & 31;
  const int r = lane & 15, hi = lane >> 4;
  const size_t m0 = (size_t)blockIdx.x * 16;
  const int n0 = blockIdx.y * (16 * NSUB);
  const _Float16* Ar = A + (m0 + r) * (size_t)lda;
  v8f acc[NSUB];
#pragma unroll
  for (int s = 0; s < NSUB; ++s) acc[s] = (v8f){};
  for (int k0 = 0; k0 < K; k0 += 32) {
    v8h a0 = *(const v8h*)(Ar + k0 + hi * 8);
    v8h a1 = *(const v8h*)(Ar + k0 + 16 + hi * 8);
    v16h av = __builtin_shufflevector(a0, a1, 0,1,2,3,4,5,6,7,8,9,10,11,12,13,14,15);
#pragma unroll
    for (int s = 0; s < NSUB; ++s) {
      const _Float16* Br = Bt + (size_t)(n0 + s * 16 + r) * ldb;
      v8h b0 = *(const v8h*)(Br + k0 + hi * 8);
      v8h b1 = *(const v8h*)(Br + k0 + 16 + hi * 8);
      v16h bv = __builtin_shufflevector(b0, b1, 0,1,2,3,4,5,6,7,8,9,10,11,12,13,14,15);
      acc[s] = __builtin_amdgcn_wmma_f32_16x16x32_f16(false, av, false, bv,
                                                      (short)0, acc[s], false, false);
    }
  }
#pragma unroll
  for (int s = 0; s < NSUB; ++s) {
    const int bcol = n0 + s * 16 + r;
    if (bcol >= N) continue;
    const float bb = bias ? bias[bcol] : 0.0f;
#pragma unroll
    for (int i = 0; i < 8; ++i) {
      size_t crow = m0 + hi * 8 + i;
      float v = acc[s][i] + bb;
      if (aux) v *= aux[crow * ldaux + auxoff + bcol];
      if (OutF) OutF[crow * ldof + bcol] = v;
      if (OutH) OutH[crow * ldoh + bcol] = (_Float16)v;
    }
  }
}

// ---------------------------------------------------------------------------
// dep() pointwise 3x3 conv as implicit-im2col WMMA GEMM, two LDS stages:
//   Block = 4 waves = 64 consecutive rows m (same b,h,py; px0..px0+63).
//   Stage 1: halo (2 br x 3 rows x 66 px x 8 ch f16 = 6.3KB), b128 loads.
//   Stage 2: im2col tile acol[(brn*64+l)*96 + k] built with COMPILE-TIME
//            LDS offsets (cpw_off(k) folds under full unroll); K pad zeroed.
//   Main loop: A fragment = 2 x ds_load_b128, same shape as global GEMM;
//   each wave computes all 5 N-subtiles (N=72 -> 80) reusing its A fragment.
// ---------------------------------------------------------------------------
__device__ constexpr int cpw_off(int k)
{
  // k = ci*9 + (dy*3+dx); halo layout [dy][px'][ci] with px' = pxl + dx
  return ((k % 9) / 3 * 66 + (k % 3)) * 8 + (k / 9);
}

__global__ __launch_bounds__(128)
void wmma_convpw_kernel(const _Float16* __restrict__ src1, const _Float16* __restrict__ src2,
                        const _Float16* __restrict__ w1t, const _Float16* __restrict__ w2t,
                        const float* __restrict__ pb1, const float* __restrict__ pb2,
                        const float* __restrict__ rpb, float* __restrict__ Out)
{
  __shared__ __align__(16) _Float16 halo[2 * 3 * 66 * 8];   // 3168 halfs
  __shared__ __align__(16) _Float16 acol[2 * 64 * 96];      // 12288 halfs
  const int tid = threadIdx.x;
  const int lane = tid & 31, wave = tid >> 5;
  const int mblk = blockIdx.x * 64;            // 64 rows per block
  const int p0 = mblk & (NPIX - 1);
  const int bh = mblk >> 14;
  const int h = bh & 7, b = bh >> 3;
  const int py = p0 >> 7, px0 = p0 & 127;

  // ---- stage 1: halo, 396 vectors of 8 halfs ----
  for (int e = tid; e < 2 * 3 * 66; e += 128) {
    int brn = e / 198;
    int rem = e - brn * 198;
    int dy = rem / 66, col = rem - dy * 66;
    int gy = py + dy - 1, gx = px0 + col - 1;
    bool ok = (gy >= 0) & (gy < HIMG) & (gx >= 0) & (gx < WIMG);
    int gyc = min(max(gy, 0), HIMG - 1);
    int gxc = min(max(gx, 0), WIMG - 1);
    const _Float16* src = brn ? src2 : src1;
    v8h v = *(const v8h*)(src + ((size_t)(b * NPIX + gyc * WIMG + gxc)) * 64 + h * 8);
    v8h z = {};
    if (!ok) v = z;
    *(v8h*)&halo[e * 8] = v;
  }
  __syncthreads();

  // ---- stage 2: im2col with compile-time offsets; thread = (brn, row) ----
  {
    int brn = tid >> 6, l = tid & 63;
    const _Float16* Hs = halo + brn * (3 * 66 * 8) + l * 8;
    _Float16* Ac = acol + (size_t)(brn * 64 + l) * 96;
#pragma unroll
    for (int k = 0; k < 72; ++k) Ac[k] = Hs[cpw_off(k)];
    v8h z = {};
    *(v8h*)(Ac + 72) = z;
    *(v8h*)(Ac + 80) = z;
    *(v8h*)(Ac + 88) = z;
  }
  __syncthreads();

  // ---- main loop ----
  const int r = lane & 15, hi = lane >> 4;
  const int pxl = wave * 16 + r;               // row within block strip
  v8f acc[5];
#pragma unroll
  for (int s = 0; s < 5; ++s) acc[s] = (v8f){};
#pragma unroll
  for (int brn = 0; brn < 2; ++brn) {
    const _Float16* Al = acol + (size_t)(brn * 64 + pxl) * 96;
    const _Float16* W  = brn ? w2t : w1t;
#pragma unroll
    for (int k0 = 0; k0 < 96; k0 += 32) {
      v8h a0 = *(const v8h*)(Al + k0 + hi * 8);
      v8h a1 = *(const v8h*)(Al + k0 + 16 + hi * 8);
      v16h av = __builtin_shufflevector(a0, a1, 0,1,2,3,4,5,6,7,8,9,10,11,12,13,14,15);
#pragma unroll
      for (int s = 0; s < 5; ++s) {
        const _Float16* Br = W + (size_t)(s * 16 + r) * 96;
        v8h b0 = *(const v8h*)(Br + k0 + hi * 8);
        v8h b1 = *(const v8h*)(Br + k0 + 16 + hi * 8);
        v16h bv = __builtin_shufflevector(b0, b1, 0,1,2,3,4,5,6,7,8,9,10,11,12,13,14,15);
        acc[s] = __builtin_amdgcn_wmma_f32_16x16x32_f16(false, av, false, bv,
                                                        (short)0, acc[s], false, false);
      }
    }
  }
#pragma unroll
  for (int s = 0; s < 5; ++s) {
    const int bcol = s * 16 + r;
    if (bcol >= 72) continue;
    const float bb = pb1[bcol] + pb2[bcol] + (rpb ? rpb[h * 9 + (bcol % 9)] : 0.0f);
#pragma unroll
    for (int i = 0; i < 8; ++i) {
      size_t crow = (size_t)mblk + wave * 16 + hi * 8 + i;
      Out[crow * 72 + bcol] = acc[s][i] + bb;
    }
  }
}

// ---------------------------------------------------------------------------
// Elementwise / small kernels
// ---------------------------------------------------------------------------
__global__ void convert_f16_kernel(const float* __restrict__ in, _Float16* __restrict__ out, int n)
{
  int i = blockIdx.x * blockDim.x + threadIdx.x;
  if (i < n) out[i] = (_Float16)in[i];
}

// pack row-major (N,K) f32 -> f16 (Npad, ldt)  [pads pre-zeroed]
__global__ void pack_nk_kernel(const float* __restrict__ w, int N, int K, int ldt,
                               _Float16* __restrict__ dst)
{
  int i = blockIdx.x * blockDim.x + threadIdx.x;
  if (i >= N * K) return;
  int n = i / K, k = i - n * K;
  dst[(size_t)n * ldt + k] = (_Float16)w[i];
}

// pack row-major (K,N) f32 -> transposed f16 (Npad, ldt)
__global__ void pack_kn_kernel(const float* __restrict__ w, int K, int N, int ldt,
                               _Float16* __restrict__ dst)
{
  int i = blockIdx.x * blockDim.x + threadIdx.x;
  if (i >= N * K) return;
  int k = i / N, n = i - k * N;
  dst[(size_t)n * ldt + k] = (_Float16)w[i];
}

// depthwise 3x3 on qkv k/v slices with both dc1/dc2 weights -> 4 f16 outputs
__global__ void dw3_qkv_kernel(const float* __restrict__ qkv,
                               const float* __restrict__ w1, const float* __restrict__ b1,
                               const float* __restrict__ w2, const float* __restrict__ b2,
                               _Float16* __restrict__ ok1, _Float16* __restrict__ ok2,
                               _Float16* __restrict__ ov1, _Float16* __restrict__ ov2)
{
  int idx = blockIdx.x * blockDim.x + threadIdx.x;
  if (idx >= MPIX * 64) return;
  int c = idx & 63, mp = idx >> 6;
  int d = c & 7, h = c >> 3;
  int p = mp & (NPIX - 1), b = mp >> 14;
  int py = p >> 7, px = p & 127;
  float sk1 = b1[d], sk2 = b2[d], sv1 = b1[d], sv2 = b2[d];
#pragma unroll
  for (int ky = 0; ky < 3; ++ky) {
    int ny = py + ky - 1;
    if (ny < 0 || ny >= HIMG) continue;
#pragma unroll
    for (int kx = 0; kx < 3; ++kx) {
      int nx = px + kx - 1;
      if (nx < 0 || nx >= WIMG) continue;
      size_t base = ((size_t)(b * NPIX + ny * WIMG + nx)) * 192 + h * 24;
      float kv = qkv[base + 8 + d];
      float vv = qkv[base + 16 + d];
      float wa = w1[d * 9 + ky * 3 + kx];
      float wb = w2[d * 9 + ky * 3 + kx];
      sk1 += kv * wa; sk2 += kv * wb;
      sv1 += vv * wa; sv2 += vv * wb;
    }
  }
  size_t o = (size_t)mp * 64 + c;
  ok1[o] = (_Float16)sk1; ok2[o] = (_Float16)sk2;
  ov1[o] = (_Float16)sv1; ov2[o] = (_Float16)sv2;
}

// attn1 = softmax_kk( sum_d q[d]*k[d*9+kk] ); writes f32 (ld 72) and f16 (ld 96)
__global__ void attn_kernel(const float* __restrict__ qkv,
                            const float* __restrict__ kout,
                            float* __restrict__ attn32, _Float16* __restrict__ attnh)
{
  int idx = blockIdx.x * blockDim.x + threadIdx.x;   // (b*HEADS+h)*NPIX + p
  if (idx >= MROWS) return;
  int p = idx & (NPIX - 1);
  int bh = idx >> 14;
  int h = bh & 7, b = bh >> 3;
  const float* kr = kout + (size_t)idx * 72;
  const float* qr = qkv + (size_t)(b * NPIX + p) * 192 + h * 24;
  float s[9];
#pragma unroll
  for (int q9 = 0; q9 < 9; ++q9) s[q9] = 0.0f;
#pragma unroll
  for (int d = 0; d < 8; ++d) {
    float q = qr[d];
#pragma unroll
    for (int q9 = 0; q9 < 9; ++q9) s[q9] += q * kr[d * 9 + q9];
  }
  float mx = s[0];
#pragma unroll
  for (int q9 = 1; q9 < 9; ++q9) mx = fmaxf(mx, s[q9]);
  float den = 0.0f;
#pragma unroll
  for (int q9 = 0; q9 < 9; ++q9) { s[q9] = expf(s[q9] - mx); den += s[q9]; }
  float inv = 1.0f / den;
  size_t mp = (size_t)(b * NPIX + p);
#pragma unroll
  for (int q9 = 0; q9 < 9; ++q9) {
    float v = s[q9] * inv;
    attn32[mp * 72 + h * 9 + q9] = v;
    attnh [mp * 96 + h * 9 + q9] = (_Float16)v;
  }
}

// generic depthwise conv, pixel-major (MPIX, ldin) f32 in, 126ch out
__global__ void dwchain_kernel(const float* __restrict__ in, int ldin, int coff,
                               int relu_in, const float* __restrict__ w,
                               const float* __restrict__ bias, int ks, int pad,
                               float* __restrict__ out)
{
  int idx = blockIdx.x * blockDim.x + threadIdx.x;
  if (idx >= MPIX * SD) return;
  int c = idx % SD, mp = idx / SD;
  int p = mp & (NPIX - 1), b = mp >> 14;
  int py = p >> 7, px = p & 127;
  float s = bias[c];
  for (int ky = 0; ky < ks; ++ky) {
    int ny = py + ky - pad;
    if (ny < 0 || ny >= HIMG) continue;
    for (int kx = 0; kx < ks; ++kx) {
      int nx = px + kx - pad;
      if (nx < 0 || nx >= WIMG) continue;
      float v = in[((size_t)(b * NPIX + ny * WIMG + nx)) * ldin + coff + c];
      if (relu_in) v = v > 0.0f ? v : 0.0f;
      s += v * w[c * ks * ks + ky * ks + kx];
    }
  }
  out[(size_t)mp * SD + c] = s;
}

// g0 = pwa * d0  -> f16 (ld 32)
__global__ void mulg0_kernel(const float* __restrict__ fused,
                             const float* __restrict__ d3,
                             _Float16* __restrict__ g0h)
{
  int idx = blockIdx.x * blockDim.x + threadIdx.x;
  if (idx >= MPIX * 18) return;
  int c = idx % 18, mp = idx / 18;
  g0h[(size_t)mp * 32 + c] =
      (_Float16)(fused[(size_t)mp * 144 + c] * d3[(size_t)mp * SD + c]);
}

// comb[mp, h*8+d] = sum_kk (g3 + attn1) * v   -> f16 (ld 64)
__global__ void combine_kernel(const float* __restrict__ g3,
                               const float* __restrict__ attn32,
                               const float* __restrict__ vout,
                               _Float16* __restrict__ combh)
{
  int idx = blockIdx.x * blockDim.x + threadIdx.x;
  if (idx >= MPIX * 64) return;
  int c = idx & 63, mp = idx >> 6;
  int d = c & 7, h = c >> 3;
  int p = mp & (NPIX - 1), b = mp >> 14;
  const float* grow = g3     + (size_t)mp * 72 + h * 9;
  const float* arow = attn32 + (size_t)mp * 72 + h * 9;
  const float* vrow = vout + ((size_t)((b * HEADS + h) * NPIX + p)) * 72 + d * 9;
  float s = 0.0f;
#pragma unroll
  for (int q9 = 0; q9 < 9; ++q9) s += (grow[q9] + arow[q9]) * vrow[q9];
  combh[(size_t)mp * 64 + c] = (_Float16)s;
}

// ---------------------------------------------------------------------------
// Orchestration
// ---------------------------------------------------------------------------
extern "C" void kernel_launch(void* const* d_in, const int* in_sizes, int n_in,
                              void* d_out, int out_size, void* d_ws, size_t ws_size,
                              hipStream_t stream)
{
  const float* x        = (const float*)d_in[0];
  const float* qkv_w    = (const float*)d_in[1];
  const float* qkv_b    = (const float*)d_in[2];
  const float* dc1_dw_w = (const float*)d_in[3];
  const float* dc1_dw_b = (const float*)d_in[4];
  const float* dc1_pw_w = (const float*)d_in[5];
  const float* dc1_pw_b = (const float*)d_in[6];
  const float* dc2_dw_w = (const float*)d_in[7];
  const float* dc2_dw_b = (const float*)d_in[8];
  const float* dc2_pw_w = (const float*)d_in[9];
  const float* dc2_pw_b = (const float*)d_in[10];
  const float* rpb      = (const float*)d_in[11];
  const float* pin_w    = (const float*)d_in[12];
  const float* pin_b    = (const float*)d_in[13];
  const float* dw7_w    = (const float*)d_in[14];
  const float* dw7_b    = (const float*)d_in[15];
  const float* dw5_w    = (const float*)d_in[16];
  const float* dw5_b    = (const float*)d_in[17];
  const float* dw3_w    = (const float*)d_in[18];
  const float* dw3_b    = (const float*)d_in[19];
  const float* pw1_w    = (const float*)d_in[20];
  const float* pw1_b    = (const float*)d_in[21];
  const float* pw2_w    = (const float*)d_in[22];
  const float* pw2_b    = (const float*)d_in[23];
  const float* pout_w   = (const float*)d_in[24];
  const float* pout_b   = (const float*)d_in[25];
  const float* proj_w   = (const float*)d_in[26];
  const float* proj_b   = (const float*)d_in[27];
  float* out = (float*)d_out;
  (void)in_sizes; (void)n_in; (void)out_size; (void)ws_size;

  // ---- workspace layout: f32 arena then f16 arena ----
  float* wsf = (float*)d_ws;
  size_t off = 0;
  auto af = [&](size_t n) { float* p = wsf + off; off += n; return p; };
  float* qkv    = af((size_t)MPIX * 192);
  float* kout   = af((size_t)MROWS * 72);
  float* vout   = af((size_t)MROWS * 72);
  float* attn32 = af((size_t)MPIX * 72);
  float* fused  = af((size_t)MPIX * 144);
  float* t1     = af((size_t)MPIX * SD);
  float* t2     = af((size_t)MPIX * SD);
  float* t3     = af((size_t)MPIX * SD);
  float* g3     = af((size_t)MPIX * 72);

  _Float16* wsh = (_Float16*)(wsf + off);
  size_t hoff = 0;
  auto ah = [&](size_t n) { _Float16* p = wsh + hoff; hoff += n; return p; };
  _Float16* xh      = ah((size_t)MPIX * 256);
  _Float16* dwk1h   = ah((size_t)MPIX * 64);
  _Float16* dwk2h   = ah((size_t)MPIX * 64);
  _Float16* dwv1h   = ah((size_t)MPIX * 64);
  _Float16* dwv2h   = ah((size_t)MPIX * 64);
  _Float16* attnh   = ah((size_t)MPIX * 96);
  _Float16* g0h     = ah((size_t)MPIX * 32);
  _Float16* g1h     = ah((size_t)MPIX * 64);
  _Float16* g2h     = ah((size_t)MPIX * 96);
  _Float16* combh   = ah((size_t)MPIX * 64);
  _Float16* qkv_wt  = ah((size_t)192 * 256);
  _Float16* pin_wt  = ah((size_t)144 * 96);
  _Float16* pw1_wt  = ah((size_t)48 * 32);
  _Float16* pw2_wt  = ah((size_t)80 * 64);
  _Float16* pout_wt = ah((size_t)80 * 96);
  _Float16* dc1_wt  = ah((size_t)80 * 96);
  _Float16* dc2_wt  = ah((size_t)80 * 96);
  _Float16* proj_wt = ah((size_t)256 * 64);

  const dim3 w32(32);

  // zero the f16 arena (establishes all K/N padding invariants)
  hipMemsetAsync(wsh, 0, hoff * sizeof(_Float16), stream);

  // operand packing / conversion
  convert_f16_kernel<<<(MPIX * 256 + 255) / 256, 256, 0, stream>>>(x, xh, MPIX * 256);
  pack_kn_kernel<<<(256 * 192 + 255) / 256, 256, 0, stream>>>(qkv_w, 256, 192, 256, qkv_wt);
  pack_nk_kernel<<<(144 * 72 + 255) / 256, 256, 0, stream>>>(pin_w, 144, 72, 96, pin_wt);
  pack_nk_kernel<<<(36 * 18 + 255) / 256, 256, 0, stream>>>(pw1_w, 36, 18, 32, pw1_wt);
  pack_nk_kernel<<<(72 * 36 + 255) / 256, 256, 0, stream>>>(pw2_w, 72, 36, 64, pw2_wt);
  pack_nk_kernel<<<(72 * 72 + 255) / 256, 256, 0, stream>>>(pout_w, 72, 72, 96, pout_wt);
  pack_nk_kernel<<<(72 * 72 + 255) / 256, 256, 0, stream>>>(dc1_pw_w, 72, 72, 96, dc1_wt);
  pack_nk_kernel<<<(72 * 72 + 255) / 256, 256, 0, stream>>>(dc2_pw_w, 72, 72, 96, dc2_wt);
  pack_kn_kernel<<<(64 * 256 + 255) / 256, 256, 0, stream>>>(proj_w, 64, 256, 64, proj_wt);

  // 1) qkv = x @ qkv_w + qkv_b          -> f32 (MPIX, 192)
  wmma_gemm_f16<4><<<dim3(MPIX / 16, 3), w32, 0, stream>>>(
      xh, 256, qkv_wt, 256, qkv_b, nullptr, 0, 0, qkv, 192, nullptr, 0, 192, 256);

  // 2) depthwise 3x3 on k/v slices (dc1 & dc2) -> 4x f16 (MPIX, 64)
  dw3_qkv_kernel<<<(MPIX * 64 + 255) / 256, 256, 0, stream>>>(
      qkv, dc1_dw_w, dc1_dw_b, dc2_dw_w, dc2_dw_b, dwk1h, dwk2h, dwv1h, dwv2h);

  // 3) dep() pointwise 3x3 (WMMA + LDS im2col); k gets +rpb
  {
    dim3 g(MROWS / 64);
    wmma_convpw_kernel<<<g, dim3(128), 0, stream>>>(dwk1h, dwk2h, dc1_wt, dc2_wt,
                                                    dc1_pw_b, dc2_pw_b, rpb, kout);
    wmma_convpw_kernel<<<g, dim3(128), 0, stream>>>(dwv1h, dwv2h, dc1_wt, dc2_wt,
                                                    dc1_pw_b, dc2_pw_b, nullptr, vout);
  }

  // 4) attn1 = softmax(q . k)           -> f32 (72) + f16 (96)
  attn_kernel<<<(MROWS + 255) / 256, 256, 0, stream>>>(qkv, kout, attn32, attnh);

  // 5) fused = attn @ pin_w + pin_b     -> f32 (MPIX, 144)
  wmma_gemm_f16<3><<<dim3(MPIX / 16, 3), w32, 0, stream>>>(
      attnh, 96, pin_wt, 96, pin_b, nullptr, 0, 0, fused, 144, nullptr, 0, 144, 96);

  // 6) depthwise 7x7 -> relu -> 5x5 -> relu -> 3x3 chain
  {
    int nb = (MPIX * SD + 255) / 256;
    dwchain_kernel<<<nb, 256, 0, stream>>>(fused, 144, 18, 0, dw7_w, dw7_b, 7, 3, t1);
    dwchain_kernel<<<nb, 256, 0, stream>>>(t1, SD, 0, 1, dw5_w, dw5_b, 5, 2, t2);
    dwchain_kernel<<<nb, 256, 0, stream>>>(t2, SD, 0, 1, dw3_w, dw3_b, 3, 1, t3);
  }

  // 7) g0 = pwa * d0 -> f16
  mulg0_kernel<<<(MPIX * 18 + 255) / 256, 256, 0, stream>>>(fused, t3, g0h);

  // 8-10) gated pointwise chain (WMMA, f16 intermediates)
  wmma_gemm_f16<3><<<dim3(MPIX / 16, 1), w32, 0, stream>>>(
      g0h, 32, pw1_wt, 32, pw1_b, t3, SD, 18, nullptr, 0, g1h, 64, 36, 32);
  wmma_gemm_f16<5><<<dim3(MPIX / 16, 1), w32, 0, stream>>>(
      g1h, 64, pw2_wt, 64, pw2_b, t3, SD, 54, nullptr, 0, g2h, 96, 72, 64);
  wmma_gemm_f16<5><<<dim3(MPIX / 16, 1), w32, 0, stream>>>(
      g2h, 96, pout_wt, 96, pout_b, nullptr, 0, 0, g3, 72, nullptr, 0, 72, 96);

  // 11) comb = sum_kk (g3 + attn1) * v  -> f16 (MPIX, 64)
  combine_kernel<<<(MPIX * 64 + 255) / 256, 256, 0, stream>>>(g3, attn32, vout, combh);

  // 12) out = comb @ proj_w + proj_b    -> f32 (B, N, 256)
  wmma_gemm_f16<4><<<dim3(MPIX / 16, 4), w32, 0, stream>>>(
      combh, 64, proj_wt, 64, proj_b, nullptr, 0, 0, out, 256, nullptr, 0, 256, 64);
}
